// GraphNNConvModel_35278861369958
// MI455X (gfx1250) — matrix operand order repaired and checked by reading, not compiled, not run on hardware
//
#include <hip/hip_runtime.h>
#include <hip/hip_bf16.h>

typedef __attribute__((ext_vector_type(16))) _Float16 v16h;
typedef __attribute__((ext_vector_type(8)))  _Float16 v8h;
typedef __attribute__((ext_vector_type(8)))  float    v8f;

#define HID 32
#define EDGE_DIM 4

// ---------------------------------------------------------------------------
// pack_w2: one-shot per layer. Builds padded, transposed f16 image of the
// bilinear-form weights in i-major K order:
//   kk = i*32 + k  (i in [0,IC), k in [0,32)) : W2''[kk,o] = w2[k, i*OC + o]
//   kk = K_CORE + i (i < IC)                  : b2[i*OC + o]   (b2 folded in)
//   else 0.  Stored column-major: W2h[o * K_PAD + kk], o in [0, OCP).
// ---------------------------------------------------------------------------
template<int IC, int OC>
__global__ void pack_w2_kernel(const float* __restrict__ w2,
                               const float* __restrict__ b2,
                               _Float16* __restrict__ W2h) {
    constexpr int K_CORE = 32 * IC;
    constexpr int K_PAD  = K_CORE + 32;
    constexpr int OCP    = ((OC + 15) / 16) * 16;
    int idx = blockIdx.x * blockDim.x + threadIdx.x;
    if (idx >= K_PAD * OCP) return;
    int o = idx / K_PAD, kk = idx % K_PAD;
    float v = 0.0f;
    if (o < OC) {
        if (kk < K_CORE) {
            int i = kk >> 5, k = kk & 31;
            v = w2[k * (IC * OC) + i * OC + o];
        } else {
            int i = kk - K_CORE;
            if (i < IC) v = b2[i * OC + o];
        }
    }
    W2h[idx] = (_Float16)v;
}

// ---------------------------------------------------------------------------
// Edge kernel: msg(E x OC) = P(E x K_PAD) @ W2''(K_PAD x OC) with
//   P[e, i*32+k] = z_e[i] * h_e[k],  P[e, K_CORE+i] = z_e[i]  (b2 rows)
// 4 waves/block, 4 tiles of 16 edges per wave (256 edges/block).
// A fragments: hfrag loaded once per tile from LDS (hi-dependent ADDRESS,
// no register selects); per chunk a = hfrag * z[c] (constant extract).
// ---------------------------------------------------------------------------
template<int IC, int OC>
__global__ __launch_bounds__(128)
void edge_wmma_kernel(const float* __restrict__ hprev, int ld, int off,
                      const int*   __restrict__ ei,      // [2*E] row0=src row1=dst
                      const float* __restrict__ ea,      // [E*4]
                      const float* __restrict__ w1,      // [4*32]
                      const float* __restrict__ b1,      // [32]
                      const _Float16* __restrict__ W2h,  // packed [OCP][K_PAD]
                      float* __restrict__ agg,           // [N*OC]
                      int E_) {
    constexpr int K_CORE = 32 * IC;
    constexpr int K_PAD  = K_CORE + 32;
    constexpr int OCP    = ((OC + 15) / 16) * 16;
    constexpr int NT     = OCP / 16;
    constexpr int WAVES  = 4;
    constexpr int TPW    = 4;             // 16-edge tiles per wave

    extern __shared__ __align__(16) char smem[];
    _Float16* Bh    = (_Float16*)smem;                                  // [OCP*K_PAD]
    _Float16* stage = (_Float16*)(smem + (size_t)OCP * K_PAD * sizeof(_Float16));

    const int wid  = threadIdx.x >> 5;
    const int lane = threadIdx.x & 31;
    _Float16* hS = stage + wid * (16 * 32 + 16 * 32);  // [16][32] edge-MLP out (f16)
    _Float16* zS = hS + 16 * 32;                       // [16][32] gathered z, zero-padded

    __builtin_prefetch(W2h, 0, 1);

    // ---- Copy packed B image into LDS (vectorized) ----
    {
        const uint4* srcv = (const uint4*)W2h;
        uint4* dstv = (uint4*)Bh;
        constexpr int NV = (K_PAD * OCP) / 8;          // halves -> uint4
        for (int idx = threadIdx.x; idx < NV; idx += 128) dstv[idx] = srcv[idx];
    }
    __syncthreads();

    const int r16 = lane & 15;
    const int hi  = lane >> 4;
    float w1a[EDGE_DIM];
#pragma unroll
    for (int a = 0; a < EDGE_DIM; ++a) w1a[a] = w1[a * HID + lane];
    const float b1l = b1[lane];

    for (int tile = 0; tile < TPW; ++tile) {
        const int e0 = blockIdx.x * (WAVES * TPW * 16) + wid * (TPW * 16) + tile * 16;

        // ---- Edge MLP: h_e = relu(ea @ w1 + b1); lane owns hidden unit k=lane ----
#pragma unroll 4
        for (int r = 0; r < 16; ++r) {
            int e = e0 + r; if (e >= E_) e = E_ - 1;   // clamp; masked by z=0
            float acc = b1l;
#pragma unroll
            for (int a = 0; a < EDGE_DIM; ++a) acc += ea[(size_t)e * EDGE_DIM + a] * w1a[a];
            hS[r * 32 + lane] = (_Float16)(acc > 0.0f ? acc : 0.0f);
        }

        // ---- Gather z = h_prev[src[e]], zero-padded to 32 cols ----
#pragma unroll 4
        for (int t = lane; t < 16 * 32; t += 32) {
            int r = t >> 5, i = lane;                  // i == lane
            int e = e0 + r;
            float v = 0.0f;
            if (i < IC && e < E_) v = hprev[(size_t)ei[e] * ld + off + i];
            zS[r * 32 + i] = (_Float16)v;
        }

        // ---- Per-lane fragments straight from LDS (hi picks the ADDRESS) ----
        // A k order chosen here (contiguous; B uses the same): lanes 0-15 elems 0-15 = K 0-15;
        // lanes 16-31 elems 0-15 = K 16-31.  k-pattern is chunk-invariant -> hfrag loaded once.
        const _Float16* hrow = hS + r16 * 32;
        const _Float16* zrow = zS + r16 * 32;
        v8h hf0 = *(const v8h*)(hrow + hi * 16);       // h[hi*16 .. +7]
        v8h hf1 = *(const v8h*)(hrow + hi * 16 + 8);   // h[hi*16+8 .. +7]
        v8h zt0 = *(const v8h*)(zrow + hi * 16);       // tail A = z (padded)
        v8h zt1 = *(const v8h*)(zrow + hi * 16 + 8);
        v16h hf, zfrag;
#pragma unroll
        for (int j = 0; j < 8; ++j) {
            hf[j] = hf0[j];    hf[j + 8] = hf1[j];
            zfrag[j] = zt0[j]; zfrag[j + 8] = zt1[j];
        }
        v16h zA0 = *(const v16h*)zrow;                 // z[0..15] for scalar extracts
        v16h zA1 = *(const v16h*)(zrow + 16);          // z[16..31]

        v8f acc[NT];
#pragma unroll
        for (int t = 0; t < NT; ++t)
#pragma unroll
            for (int q = 0; q < 8; ++q) acc[t][q] = 0.0f;

        const _Float16* Bcol[NT];
#pragma unroll
        for (int t = 0; t < NT; ++t) Bcol[t] = Bh + (size_t)(t * 16 + r16) * K_PAD;

        // ---- Core K chunks: chunk c has z-index i == c (constant extract) ----
#pragma unroll
        for (int c = 0; c < IC; ++c) {
            const _Float16 zc = (c < 16) ? zA0[c] : zA1[c - 16];
            v16h a;
#pragma unroll
            for (int j = 0; j < 16; ++j) a[j] = hf[j] * zc;
#pragma unroll
            for (int t = 0; t < NT; ++t) {
                v16h b = *(const v16h*)(Bcol[t] + c * 32 + hi * 16);
                acc[t] = __builtin_amdgcn_wmma_f32_16x16x32_f16(
                    false, a, false, b, (short)0, acc[t], false, false);
            }
        }

        // ---- Tail chunk (b2 rows): A = zfrag directly ----
#pragma unroll
        for (int t = 0; t < NT; ++t) {
            v16h b = *(const v16h*)(Bcol[t] + K_CORE + hi * 16);
            acc[t] = __builtin_amdgcn_wmma_f32_16x16x32_f16(
                false, zfrag, false, b, (short)0, acc[t], false, false);
        }

        // ---- Scatter D tile: segment-sum over dst via f32 global atomics ----
        // lanes 0-15: acc[r] = D[M=r][N=r16]; lanes 16-31: D[M=r+8][N=r16]
#pragma unroll
        for (int t = 0; t < NT; ++t) {
            const int o = t * 16 + r16;
#pragma unroll
            for (int r = 0; r < 8; ++r) {
                int e = e0 + r + hi * 8;
                if (e < E_ && o < OC)
                    atomicAdd(&agg[(size_t)ei[E_ + e] * OC + o], acc[t][r]);
            }
        }
    }
}

// ---------------------------------------------------------------------------
// Node kernel: agg[n,o] = bias[o] + sum_i h_prev[n,i] * root[i,o]
// ---------------------------------------------------------------------------
template<int IC, int OC>
__global__ void node_root_kernel(const float* __restrict__ hprev, int ld, int off,
                                 const float* __restrict__ root,
                                 const float* __restrict__ bias,
                                 float* __restrict__ agg, int Nn) {
    int idx = blockIdx.x * blockDim.x + threadIdx.x;
    if (idx >= Nn * OC) return;
    int n = idx / OC, o = idx % OC;
    const float* hr = hprev + (size_t)n * ld + off;
    float acc = bias[o];
#pragma unroll
    for (int i = 0; i < IC; ++i) acc += hr[i] * root[i * OC + o];
    agg[idx] = acc;
}

__global__ void relu_kernel(const float* __restrict__ in, float* __restrict__ out, int n) {
    int idx = blockIdx.x * blockDim.x + threadIdx.x;
    if (idx < n) out[idx] = fmaxf(in[idx], 0.0f);
}

__global__ void copy_kernel(const float* __restrict__ in, float* __restrict__ out, int n) {
    int idx = blockIdx.x * blockDim.x + threadIdx.x;
    if (idx < n) out[idx] = in[idx];
}

// ---------------------------------------------------------------------------
template<int IC, int OC>
static size_t edge_smem_bytes() {
    const int K_PAD = 32 * IC + 32;
    const int OCP   = ((OC + 15) / 16) * 16;
    return (size_t)K_PAD * OCP * sizeof(_Float16)
         + 4u * (16 * 32 + 16 * 32) * sizeof(_Float16);
}

static inline int cdiv(int a, int b) { return (a + b - 1) / b; }

template<int IC, int OC>
static void run_layer(const float* hprev, int ld, int off,
                      const int* ei, const float* ea,
                      const float* w1, const float* b1,
                      const float* w2, const float* b2,
                      const float* root, const float* bias,
                      _Float16* W2h, float* agg,
                      int N, int E, hipStream_t stream) {
    const int K_PAD = 32 * IC + 32;
    const int OCP   = ((OC + 15) / 16) * 16;
    pack_w2_kernel<IC, OC><<<cdiv(K_PAD * OCP, 256), 256, 0, stream>>>(w2, b2, W2h);
    node_root_kernel<IC, OC><<<cdiv(N * OC, 256), 256, 0, stream>>>(
        hprev, ld, off, root, bias, agg, N);
    edge_wmma_kernel<IC, OC><<<cdiv(E, 256), 128, edge_smem_bytes<IC, OC>(), stream>>>(
        hprev, ld, off, ei, ea, w1, b1, W2h, agg, E);
}

extern "C" void kernel_launch(void* const* d_in, const int* in_sizes, int n_in,
                              void* d_out, int out_size, void* d_ws, size_t ws_size,
                              hipStream_t stream) {
    const float* x  = (const float*)d_in[0];
    const int*   ei = (const int*)  d_in[1];
    const float* ea = (const float*)d_in[2];
    auto P = [&](int layer, int j) { return (const float*)d_in[3 + layer * 6 + j]; };
    // per layer: 0=w1 1=b1 2=w2 3=b2 4=root 5=bias

    const int N = in_sizes[0] / 10;
    const int E = in_sizes[1] / 2;

    float* agg = (float*)d_ws;                    // N*32
    float* hA  = agg + (size_t)N * 32;            // N*32
    float* hB  = hA  + (size_t)N * 32;            // N*32
    _Float16* W2h = (_Float16*)(hB + (size_t)N * 32);  // up to (32*32+32)*32 halves
    float* out = (float*)d_out;                   // N*2

    // Layer 0: IC=6 (x cols 4..9), OC=32
    run_layer<6, 32>(x, 10, 4, ei, ea, P(0,0), P(0,1), P(0,2), P(0,3), P(0,4), P(0,5),
                     W2h, agg, N, E, stream);
    relu_kernel<<<cdiv(N * 32, 256), 256, 0, stream>>>(agg, hA, N * 32);

    // Layer 1: IC=32, OC=32
    run_layer<32, 32>(hA, 32, 0, ei, ea, P(1,0), P(1,1), P(1,2), P(1,3), P(1,4), P(1,5),
                      W2h, agg, N, E, stream);
    relu_kernel<<<cdiv(N * 32, 256), 256, 0, stream>>>(agg, hB, N * 32);

    // Layer 2: IC=32, OC=32
    run_layer<32, 32>(hB, 32, 0, ei, ea, P(2,0), P(2,1), P(2,2), P(2,3), P(2,4), P(2,5),
                      W2h, agg, N, E, stream);
    relu_kernel<<<cdiv(N * 32, 256), 256, 0, stream>>>(agg, hA, N * 32);

    // Layer 3: IC=32, OC=2 (no relu)
    run_layer<32, 2>(hA, 32, 0, ei, ea, P(3,0), P(3,1), P(3,2), P(3,3), P(3,4), P(3,5),
                     W2h, agg, N, E, stream);
    copy_kernel<<<cdiv(N * 2, 256), 256, 0, stream>>>(agg, out, N * 2);
}
